// PointNetFeaturePropagation_58798102282445
// MI455X (gfx1250) — matrix operand order repaired and checked by reading, not compiled, not run on hardware
//
#include <hip/hip_runtime.h>

// ---------------------------------------------------------------------------
// Types for CDNA5 WMMA (wave32)
// ---------------------------------------------------------------------------
typedef __bf16 bf16x16 __attribute__((ext_vector_type(16)));
typedef __bf16 bf16x8  __attribute__((ext_vector_type(8)));
typedef float  f32x8   __attribute__((ext_vector_type(8)));

__device__ __forceinline__ float gelu_exact(float x) {
    return 0.5f * x * (1.0f + erff(x * 0.70710678118654752440f));
}

__device__ __forceinline__ unsigned short f2bf(float f) {
    union { float f; unsigned int u; } v; v.f = f;
    unsigned int u = v.u;
    if ((u & 0x7fffffffu) > 0x7f800000u) return (unsigned short)((u >> 16) | 0x40); // quiet NaN
    u += 0x7fffu + ((u >> 16) & 1u);   // round-to-nearest-even
    return (unsigned short)(u >> 16);
}

// ---------------------------------------------------------------------------
// 1) time MLP: t0[b,c] = GELU(t_embed[b]) @ tW0^T + tb0 ; t1[b,o] likewise.
//    (per-channel constants tb* kept for fidelity; conv biases cb* are
//     dropped everywhere because training-mode BN subtracts them exactly)
// ---------------------------------------------------------------------------
__global__ __launch_bounds__(384) void time_mlp_kernel(
    const float* __restrict__ t_embed, const float* __restrict__ tW0,
    const float* __restrict__ tb0, const float* __restrict__ tW1,
    const float* __restrict__ tb1, float* __restrict__ t0, float* __restrict__ t1)
{
    __shared__ float ge[128];
    const int b = blockIdx.x;
    const int tid = threadIdx.x;
    if (tid < 128) ge[tid] = gelu_exact(t_embed[b * 128 + tid]);
    __syncthreads();
    if (tid < 384) {
        float s = tb0[tid];
        #pragma unroll 8
        for (int t = 0; t < 128; ++t) s += ge[t] * tW0[tid * 128 + t];
        t0[b * 384 + tid] = s;
    }
    if (tid < 256) {
        float s = tb1[tid];
        #pragma unroll 8
        for (int t = 0; t < 128; ++t) s += ge[t] * tW1[tid * 128 + t];
        t1[b * 256 + tid] = s;
    }
}

// ---------------------------------------------------------------------------
// 2) convert conv weights fp32 -> bf16
// ---------------------------------------------------------------------------
__global__ __launch_bounds__(256) void cvt_weights_kernel(
    const float* __restrict__ W0, const float* __restrict__ W1,
    unsigned short* __restrict__ W0b, unsigned short* __restrict__ W1b)
{
    const int i = blockIdx.x * 256 + threadIdx.x;
    if (i < 256 * 384) W0b[i] = f2bf(W0[i]);
    if (i < 128 * 256) W1b[i] = f2bf(W1[i]);
}

// ---------------------------------------------------------------------------
// 3) 3-NN + inverse-distance weights.  xyz2[b] cached in LDS (24 KB).
// ---------------------------------------------------------------------------
__global__ __launch_bounds__(256) void knn_kernel(
    const float* __restrict__ xyz1, const float* __restrict__ xyz2,
    int* __restrict__ idx, float* __restrict__ wgt, int Np, int S)
{
    __shared__ float sx[2048], sy[2048], sz[2048];
    const int b = blockIdx.y;
    for (int s = threadIdx.x; s < S; s += 256) {
        const float* p = xyz2 + ((size_t)b * S + s) * 3;
        sx[s] = p[0]; sy[s] = p[1]; sz[s] = p[2];
    }
    __syncthreads();
    const int n = blockIdx.x * 256 + threadIdx.x;
    const float* q = xyz1 + ((size_t)b * Np + n) * 3;
    const float qx = q[0], qy = q[1], qz = q[2];
    float d0 = 3.4e38f, d1 = 3.4e38f, d2 = 3.4e38f;
    int   i0 = 0, i1 = 0, i2 = 0;
    for (int s = 0; s < S; ++s) {
        const float dx = qx - sx[s], dy = qy - sy[s], dz = qz - sz[s];
        const float d = dx * dx + dy * dy + dz * dz;
        if (d < d2) {
            if (d < d1) {
                d2 = d1; i2 = i1;
                if (d < d0) { d1 = d0; i1 = i0; d0 = d; i0 = s; }
                else        { d1 = d;  i1 = s; }
            } else { d2 = d; i2 = s; }
        }
    }
    float w0 = 1.0f / (d0 + 1e-8f);
    float w1 = 1.0f / (d1 + 1e-8f);
    float w2 = 1.0f / (d2 + 1e-8f);
    const float inv = 1.0f / (w0 + w1 + w2);
    const size_t o = ((size_t)b * Np + n) * 3;
    idx[o] = i0; idx[o + 1] = i1; idx[o + 2] = i2;
    wgt[o] = w0 * inv; wgt[o + 1] = w1 * inv; wgt[o + 2] = w2 * inv;
}

// ---------------------------------------------------------------------------
// 4) transpose points2 (B,C2,S) -> (B,S,C2) so gathers are channel-coalesced
// ---------------------------------------------------------------------------
__global__ __launch_bounds__(256) void transpose_p2_kernel(
    const float* __restrict__ p2, float* __restrict__ p2t, int C, int S)
{
    __shared__ float tile[32][33];
    const int b = blockIdx.z;
    const int c0 = blockIdx.y * 32, s0 = blockIdx.x * 32;
    const int tx = threadIdx.x & 31, ty = threadIdx.x >> 5;
    #pragma unroll
    for (int i = 0; i < 4; ++i)
        tile[ty + i * 8][tx] = p2[((size_t)b * C + c0 + ty + i * 8) * S + s0 + tx];
    __syncthreads();
    #pragma unroll
    for (int i = 0; i < 4; ++i)
        p2t[((size_t)b * S + s0 + ty + i * 8) * C + c0 + tx] = tile[tx][ty + i * 8];
}

// ---------------------------------------------------------------------------
// 5) X[b,n,c] = points1[b,c,n] + t0[b,c]  (c < 128), bf16, LDS transpose
// ---------------------------------------------------------------------------
__global__ __launch_bounds__(256) void build_x_low_kernel(
    const float* __restrict__ p1, const float* __restrict__ t0,
    unsigned short* __restrict__ Xb, int C1, int Np, int K)
{
    __shared__ float tile[32][33];
    const int b = blockIdx.z;
    const int c0 = blockIdx.y * 32, n0 = blockIdx.x * 32;
    const int tx = threadIdx.x & 31, ty = threadIdx.x >> 5;
    #pragma unroll
    for (int i = 0; i < 4; ++i) {
        const int c = c0 + ty + i * 8;
        tile[ty + i * 8][tx] = p1[((size_t)b * C1 + c) * Np + n0 + tx] + t0[b * 384 + c];
    }
    __syncthreads();
    #pragma unroll
    for (int i = 0; i < 4; ++i) {
        const int n = n0 + ty + i * 8;
        Xb[((size_t)b * Np + n) * K + c0 + tx] = f2bf(tile[tx][ty + i * 8]);
    }
}

// ---------------------------------------------------------------------------
// 6) X[b,n,128+c] = sum_k w_k * p2t[b, idx_k, c] + t0[b,128+c]  (gather)
// ---------------------------------------------------------------------------
__global__ __launch_bounds__(256) void build_x_interp_kernel(
    const float* __restrict__ p2t, const int* __restrict__ idx,
    const float* __restrict__ wgt, const float* __restrict__ t0,
    unsigned short* __restrict__ Xb, int C2, int S, int Np, int K, int Koff)
{
    const int b = blockIdx.y;
    const int c = threadIdx.x;           // 0..255
    const float* base = p2t + (size_t)b * S * C2;
    const float tadd = t0[b * 384 + Koff + c];
    #pragma unroll
    for (int jn = 0; jn < 4; ++jn) {
        const int n = blockIdx.x * 4 + jn;
        const size_t o3 = ((size_t)b * Np + n) * 3;
        const int   i0 = idx[o3], i1 = idx[o3 + 1], i2 = idx[o3 + 2];
        const float w0 = wgt[o3], w1 = wgt[o3 + 1], w2 = wgt[o3 + 2];
        const float v = w0 * base[(size_t)i0 * C2 + c]
                      + w1 * base[(size_t)i1 * C2 + c]
                      + w2 * base[(size_t)i2 * C2 + c] + tadd;
        Xb[((size_t)b * Np + n) * K + Koff + c] = f2bf(v);
    }
}

// ---------------------------------------------------------------------------
// 7) WMMA bf16 GEMM:  Y[b,m,n] = sum_k W[m,k] * X[b,n,k]   (fp32 accumulate)
//    W: (M,K) bf16 row-major. X: (B,Np,K) bf16, K contiguous.
//    Workgroup = 8 waves as 2(M) x 4(N); wave tile 16 x 64; K step 32.
//    A operand: contiguous k = 16*hi + s (one 16-elt run per lane).
//    B operand: same contiguous k = 16*hi + s as A (one 16-elt run per lane).
// ---------------------------------------------------------------------------
__global__ __launch_bounds__(256) void gemm_wmma_bf16_kernel(
    const unsigned short* __restrict__ Wb, const unsigned short* __restrict__ Xb,
    float* __restrict__ Y, int M, int K, int Np)
{
    const int b    = blockIdx.z;
    const int lane = threadIdx.x & 31;
    const int wave = threadIdx.x >> 5;
    const int lo   = lane & 15;
    const int hi   = lane >> 4;
    const int m_base = blockIdx.y * 32 + (wave & 1) * 16;
    const int n_base = blockIdx.x * 256 + (wave >> 1) * 64;

    const unsigned short* wrow  = Wb + (size_t)(m_base + lo) * K;
    const unsigned short* xbase = Xb + (size_t)b * Np * K;
    const unsigned short* xr0 = xbase + (size_t)(n_base +  0 + lo) * K;
    const unsigned short* xr1 = xbase + (size_t)(n_base + 16 + lo) * K;
    const unsigned short* xr2 = xbase + (size_t)(n_base + 32 + lo) * K;
    const unsigned short* xr3 = xbase + (size_t)(n_base + 48 + lo) * K;

    f32x8 acc0 = {}, acc1 = {}, acc2 = {}, acc3 = {};

    for (int k0 = 0; k0 < K; k0 += 32) {
        // A: lane<16 -> K {k0..k0+15}; lane>=16 -> K {k0+16..k0+31}
        const bf16x16 a = *reinterpret_cast<const bf16x16*>(wrow + k0 + hi * 16);

        // B: lane<16 -> K {k0..k0+15}; lane>=16 -> K {k0+16..k0+31}, col = lo
        const int koff = k0 + hi * 16;
        const bf16x16 b0 = *reinterpret_cast<const bf16x16*>(xr0 + koff);
        const bf16x16 b1 = *reinterpret_cast<const bf16x16*>(xr1 + koff);
        const bf16x16 b2 = *reinterpret_cast<const bf16x16*>(xr2 + koff);
        const bf16x16 b3 = *reinterpret_cast<const bf16x16*>(xr3 + koff);

        if (k0 + 32 < K) {                       // global_prefetch next K step
            __builtin_prefetch((const void*)(xr0 + koff + 32), 0, 1);
            __builtin_prefetch((const void*)(xr2 + koff + 32), 0, 1);
        }

        acc0 = __builtin_amdgcn_wmma_f32_16x16x32_bf16(false, a, false, b0, (short)0, acc0, false, false);
        acc1 = __builtin_amdgcn_wmma_f32_16x16x32_bf16(false, a, false, b1, (short)0, acc1, false, false);
        acc2 = __builtin_amdgcn_wmma_f32_16x16x32_bf16(false, a, false, b2, (short)0, acc2, false, false);
        acc3 = __builtin_amdgcn_wmma_f32_16x16x32_bf16(false, a, false, b3, (short)0, acc3, false, false);
    }

    // D layout: VGPR r -> M = m_base + r + 8*hi, N = n_base + 16*j + lo
    float* yrow = Y + ((size_t)b * M + m_base + hi * 8) * Np + n_base + lo;
    #pragma unroll
    for (int r = 0; r < 8; ++r) {
        float* p = yrow + (size_t)r * Np;
        p[0]  = acc0[r];
        p[16] = acc1[r];
        p[32] = acc2[r];
        p[48] = acc3[r];
    }
}

// ---------------------------------------------------------------------------
// 8) per-channel BN stats: mean + rsqrt(var + eps) over (B, Np)
// ---------------------------------------------------------------------------
__global__ __launch_bounds__(256) void bn_stats_kernel(
    const float* __restrict__ Y, float2* __restrict__ stats, int M, int Np, int B)
{
    __shared__ float rs[256], rss[256];
    const int o = blockIdx.x;
    const int tid = threadIdx.x;
    float s = 0.f, ss = 0.f;
    for (int b = 0; b < B; ++b) {
        const float* p = Y + ((size_t)b * M + o) * Np;
        for (int n = tid; n < Np; n += 256) { const float v = p[n]; s += v; ss += v * v; }
    }
    rs[tid] = s; rss[tid] = ss;
    __syncthreads();
    for (int d = 128; d > 0; d >>= 1) {
        if (tid < d) { rs[tid] += rs[tid + d]; rss[tid] += rss[tid + d]; }
        __syncthreads();
    }
    if (tid == 0) {
        const float inv  = 1.0f / (float)(B * Np);
        const float mean = rs[0] * inv;
        const float var  = rss[0] * inv - mean * mean;
        stats[o] = make_float2(mean, rsqrtf(var + 1e-5f));
    }
}

// ---------------------------------------------------------------------------
// 9) x1[b,n,o] = bf16( GELU(BN(Y[b,o,n])) + t1[b,o] )   (LDS transpose)
// ---------------------------------------------------------------------------
__global__ __launch_bounds__(256) void bn_act_transpose_kernel(
    const float* __restrict__ Y, const float2* __restrict__ stats,
    const float* __restrict__ gamma, const float* __restrict__ beta,
    const float* __restrict__ tvec, unsigned short* __restrict__ Xout, int M, int Np)
{
    __shared__ float tile[32][33];
    const int b = blockIdx.z;
    const int o0 = blockIdx.y * 32, n0 = blockIdx.x * 32;
    const int tx = threadIdx.x & 31, ty = threadIdx.x >> 5;
    #pragma unroll
    for (int i = 0; i < 4; ++i) {
        const int o = o0 + ty + i * 8;
        const float2 st = stats[o];
        float v = Y[((size_t)b * M + o) * Np + n0 + tx];
        v = (v - st.x) * st.y * gamma[o] + beta[o];
        tile[ty + i * 8][tx] = gelu_exact(v) + tvec[b * M + o];
    }
    __syncthreads();
    #pragma unroll
    for (int i = 0; i < 4; ++i) {
        const int n = n0 + ty + i * 8;
        Xout[((size_t)b * Np + n) * M + o0 + tx] = f2bf(tile[tx][ty + i * 8]);
    }
}

// ---------------------------------------------------------------------------
// 10) out[b,o,n] = GELU(BN(Y1[b,o,n]))   (elementwise, fp32 out)
// ---------------------------------------------------------------------------
__global__ __launch_bounds__(256) void bn_act_out_kernel(
    const float* __restrict__ Y, const float2* __restrict__ stats,
    const float* __restrict__ gamma, const float* __restrict__ beta,
    float* __restrict__ out, int M, int Np, size_t total)
{
    const size_t i = (size_t)blockIdx.x * 256 + threadIdx.x;
    if (i >= total) return;
    const int o = (int)((i / (size_t)Np) % (size_t)M);
    const float2 st = stats[o];
    const float v = (Y[i] - st.x) * st.y * gamma[o] + beta[o];
    out[i] = gelu_exact(v);
}

// ---------------------------------------------------------------------------
// launch
// ---------------------------------------------------------------------------
extern "C" void kernel_launch(void* const* d_in, const int* in_sizes, int n_in,
                              void* d_out, int out_size, void* d_ws, size_t ws_size,
                              hipStream_t stream)
{
    const float* xyz1    = (const float*)d_in[0];
    const float* xyz2    = (const float*)d_in[1];
    const float* points1 = (const float*)d_in[2];
    const float* points2 = (const float*)d_in[3];
    const float* t_embed = (const float*)d_in[4];
    const float* tW0     = (const float*)d_in[5];
    const float* tb0     = (const float*)d_in[6];
    const float* cW0     = (const float*)d_in[7];
    // cb0 (d_in[8]) dropped: exactly cancelled by training-mode BN mean-subtract
    const float* g0      = (const float*)d_in[9];
    const float* beta0   = (const float*)d_in[10];
    const float* tW1     = (const float*)d_in[11];
    const float* tb1     = (const float*)d_in[12];
    const float* cW1     = (const float*)d_in[13];
    // cb1 (d_in[14]) dropped for the same reason
    const float* g1      = (const float*)d_in[15];
    const float* beta1   = (const float*)d_in[16];
    float* out = (float*)d_out;

    constexpr int B = 8, Np = 8192, S = 2048, C1 = 128, C2 = 256;
    constexpr int K0 = 384, H0 = 256, H1 = 128;

    char* ws = (char*)d_ws;
    size_t off = 0;
    auto take = [&](size_t bytes) -> char* {
        char* p = ws + off;
        off = (off + bytes + 255) & ~(size_t)255;
        return p;
    };
    float*          t0   = (float*)take((size_t)B * K0 * 4);
    float*          t1   = (float*)take((size_t)B * H0 * 4);
    unsigned short* W0b  = (unsigned short*)take((size_t)H0 * K0 * 2);
    unsigned short* W1b  = (unsigned short*)take((size_t)H1 * H0 * 2);
    float2*         st0  = (float2*)take((size_t)H0 * 8);
    float2*         st1  = (float2*)take((size_t)H1 * 8);
    int*            idx  = (int*)take((size_t)B * Np * 3 * 4);
    float*          wgt  = (float*)take((size_t)B * Np * 3 * 4);
    float*          p2t  = (float*)take((size_t)B * S * C2 * 4);
    unsigned short* Xb   = (unsigned short*)take((size_t)B * Np * K0 * 2);  // 50.3 MB
    float*          Y    = (float*)take((size_t)B * H0 * Np * 4);           // 67.1 MB
    unsigned short* X1b  = Xb;   // reuse: Xb dead after GEMM0
    float*          Y1   = Y;    // reuse: Y0 dead after bn_act_transpose

    time_mlp_kernel<<<B, 384, 0, stream>>>(t_embed, tW0, tb0, tW1, tb1, t0, t1);
    cvt_weights_kernel<<<(H0 * K0 + 255) / 256, 256, 0, stream>>>(cW0, cW1, W0b, W1b);
    knn_kernel<<<dim3(Np / 256, B), 256, 0, stream>>>(xyz1, xyz2, idx, wgt, Np, S);
    transpose_p2_kernel<<<dim3(S / 32, C2 / 32, B), 256, 0, stream>>>(points2, p2t, C2, S);
    build_x_low_kernel<<<dim3(Np / 32, C1 / 32, B), 256, 0, stream>>>(points1, t0, Xb, C1, Np, K0);
    build_x_interp_kernel<<<dim3(Np / 4, B), 256, 0, stream>>>(p2t, idx, wgt, t0, Xb, C2, S, Np, K0, C1);

    gemm_wmma_bf16_kernel<<<dim3(Np / 256, H0 / 32, B), 256, 0, stream>>>(W0b, Xb, Y, H0, K0, Np);
    bn_stats_kernel<<<H0, 256, 0, stream>>>(Y, st0, H0, Np, B);
    bn_act_transpose_kernel<<<dim3(Np / 32, H0 / 32, B), 256, 0, stream>>>(Y, st0, g0, beta0, t1, X1b, H0, Np);

    gemm_wmma_bf16_kernel<<<dim3(Np / 256, H1 / 32, B), 256, 0, stream>>>(W1b, X1b, Y1, H1, H0, Np);
    bn_stats_kernel<<<H1, 256, 0, stream>>>(Y1, st1, H1, Np, B);
    const size_t total = (size_t)B * H1 * Np;
    bn_act_out_kernel<<<(unsigned)((total + 255) / 256), 256, 0, stream>>>(Y1, st1, g1, beta1, out, H1, Np, total);
}
